// PointEdgeSegNet_62302795596750
// MI455X (gfx1250) — compile-verified, matrix operand order checked by reading, not run on hardware
//
#include <hip/hip_runtime.h>

// ---------------------------------------------------------------------------
// CDNA5 (gfx1250) implementation of PointEdgeSegNet forward pass.
// All dense layers run through v_wmma_f32_16x16x32_f16 (wave32 WMMA).
// Weights are pre-transposed to N-major so both WMMA operands load as
// two contiguous global_load_b128 per fragment.
// ---------------------------------------------------------------------------

typedef __attribute__((ext_vector_type(16))) _Float16 v16h;
typedef __attribute__((ext_vector_type(8)))  _Float16 v8h;
typedef __attribute__((ext_vector_type(8)))  float    v8f;

#define DIVUP(a, b) (((a) + (b) - 1) / (b))
static constexpr int B = 4;

// Load one 16-half WMMA fragment chunk pair for this lane.
// Fragment layout (CDNA5 ISA 16-bit A 16x32): lane = half*16 + r;
// vector elems [0..7]  = row[k0 + half*8 + 0..7]
// vector elems [8..15] = row[k0 + 16 + half*8 + 0..7]
__device__ inline v16h load_frag(const _Float16* __restrict__ row, int k0, int half)
{
    const v8h lo = *(const v8h*)(row + k0 + half * 8);
    const v8h hi = *(const v8h*)(row + k0 + 16 + half * 8);
    return __builtin_shufflevector(lo, hi, 0, 1, 2, 3, 4, 5, 6, 7,
                                           8, 9, 10, 11, 12, 13, 14, 15);
}

// ---------------------------------------------------------------------------
// WMMA GEMM: C[M x N] (f32) = A[M x Kp] (f16) @ Wt[N x Kp]^T (f16) + bias[N]
// Kp % 32 == 0, N % (16*NT) == 0, M % 16 == 0.
// One wave -> 16 x (16*NT) C strip: one A fragment shared by NT accumulators.
// ---------------------------------------------------------------------------
template <int NT>
__global__ __launch_bounds__(128)
void gemm_f16_wmma(const _Float16* __restrict__ A,
                   const _Float16* __restrict__ Wt,
                   const float* __restrict__ bias,
                   float* __restrict__ C,
                   int M, int Kp, int N)
{
    const int wave = threadIdx.x >> 5;
    const int lane = threadIdx.x & 31;
    const int tm   = blockIdx.x * 4 + wave;   // 16-row tile index (wave uniform)
    if (tm * 16 >= M) return;                 // uniform per wave -> EXEC stays full

    const int half = lane >> 4;
    const int mr   = lane & 15;
    const int n0   = blockIdx.y * (16 * NT);

    const _Float16* __restrict__ Arow = A + (size_t)(tm * 16 + mr) * Kp;
    const _Float16* Wrow[NT];
#pragma unroll
    for (int s = 0; s < NT; ++s)
        Wrow[s] = Wt + (size_t)(n0 + s * 16 + mr) * Kp;

    v8f acc[NT];
#pragma unroll
    for (int s = 0; s < NT; ++s) acc[s] = {};

    for (int k0 = 0; k0 < Kp; k0 += 32) {
        __builtin_prefetch(Arow + k0 + 128, 0, 1);
        const v16h a = load_frag(Arow, k0, half);
#pragma unroll
        for (int s = 0; s < NT; ++s) {
            const v16h b = load_frag(Wrow[s], k0, half);
            acc[s] = __builtin_amdgcn_wmma_f32_16x16x32_f16(
                /*neg_a=*/false, a, /*neg_b=*/false, b,
                /*c_mod=*/(short)0, acc[s], /*reuse_a=*/false, /*reuse_b=*/false);
        }
    }

#pragma unroll
    for (int s = 0; s < NT; ++s) {
        const int n  = n0 + s * 16 + mr;
        const float bv = bias[n];
#pragma unroll
        for (int r = 0; r < 8; ++r) {
            const int m = tm * 16 + half * 8 + r;
            C[(size_t)m * N + n] = acc[s][r] + bv;
        }
    }
}

// ---------------------------------------------------------------------------
// Brute-force KNN, LDS-tiled.  One thread per query, top-K by insertion.
// ---------------------------------------------------------------------------
template <int KNN, bool EXCL>
__global__ void knn_kernel(const float* __restrict__ posq,
                           const float* __restrict__ poss,
                           int Nq, int Ns,
                           int* __restrict__ idx_out,
                           float* __restrict__ d2_out)
{
    const int b = blockIdx.y;
    const int q = blockIdx.x * blockDim.x + threadIdx.x;
    __shared__ float sp[128][3];

    float qx = 0.f, qy = 0.f, qz = 0.f;
    const bool valid = q < Nq;
    if (valid) {
        const float* pq = posq + ((size_t)b * Nq + q) * 3;
        qx = pq[0]; qy = pq[1]; qz = pq[2];
    }
    float bd[KNN]; int bi[KNN];
#pragma unroll
    for (int i = 0; i < KNN; ++i) { bd[i] = 3.4e38f; bi[i] = 0; }

    for (int s0 = 0; s0 < Ns; s0 += 128) {
        const int tile = min(128, Ns - s0);
        __syncthreads();
        for (int t = threadIdx.x; t < tile; t += blockDim.x) {
            const float* ps = poss + ((size_t)b * Ns + s0 + t) * 3;
            sp[t][0] = ps[0]; sp[t][1] = ps[1]; sp[t][2] = ps[2];
        }
        __syncthreads();
        if (valid) {
            for (int t = 0; t < tile; ++t) {
                const int s = s0 + t;
                const float dx = qx - sp[t][0], dy = qy - sp[t][1], dz = qz - sp[t][2];
                float d2 = dx * dx + dy * dy + dz * dz;
                if (EXCL && s == q) d2 += 1e10f;
                if (d2 < bd[KNN - 1]) {
                    int j = KNN - 1;
                    while (j > 0 && bd[j - 1] > d2) {
                        bd[j] = bd[j - 1]; bi[j] = bi[j - 1]; --j;
                    }
                    bd[j] = d2; bi[j] = s;
                }
            }
        }
    }
    if (valid) {
        for (int i = 0; i < KNN; ++i) {
            idx_out[((size_t)b * Nq + q) * KNN + i] = bi[i];
            if (d2_out) d2_out[((size_t)b * Nq + q) * KNN + i] = fmaxf(bd[i], 0.f);
        }
    }
}

// ---------------------------------------------------------------------------
// Farthest point sampling: one workgroup per batch, dist array in LDS.
// ---------------------------------------------------------------------------
__global__ void fps_kernel(const float* __restrict__ pos, int N, int n_out,
                           int* __restrict__ out_idx)
{
    const int b = blockIdx.x;
    const int t = threadIdx.x;
    const int T = blockDim.x;
    __shared__ float dist[4096];
    __shared__ float rv[256];
    __shared__ int   ri[256];
    __shared__ float cur[3];

    const float* P = pos + (size_t)b * N * 3;
    if (t == 0) {
        cur[0] = P[0]; cur[1] = P[1]; cur[2] = P[2];
        out_idx[(size_t)b * n_out] = 0;
    }
    __syncthreads();
    for (int i = t; i < N; i += T) {
        const float dx = P[i * 3] - cur[0], dy = P[i * 3 + 1] - cur[1], dz = P[i * 3 + 2] - cur[2];
        dist[i] = dx * dx + dy * dy + dz * dz;
    }
    __syncthreads();
    for (int j = 1; j < n_out; ++j) {
        float bv = -1.f; int bidx = 0x7fffffff;
        for (int i = t; i < N; i += T) {
            const float v = dist[i];
            if (v > bv) { bv = v; bidx = i; }
        }
        rv[t] = bv; ri[t] = bidx;
        __syncthreads();
        for (int s = T >> 1; s > 0; s >>= 1) {
            if (t < s) {
                if (rv[t + s] > rv[t] || (rv[t + s] == rv[t] && ri[t + s] < ri[t])) {
                    rv[t] = rv[t + s]; ri[t] = ri[t + s];
                }
            }
            __syncthreads();
        }
        if (t == 0) {
            const int ii = ri[0];
            out_idx[(size_t)b * n_out + j] = ii;
            cur[0] = P[ii * 3]; cur[1] = P[ii * 3 + 1]; cur[2] = P[ii * 3 + 2];
        }
        __syncthreads();
        for (int i = t; i < N; i += T) {
            const float dx = P[i * 3] - cur[0], dy = P[i * 3 + 1] - cur[1], dz = P[i * 3 + 2] - cur[2];
            const float nd = dx * dx + dy * dy + dz * dz;
            if (nd < dist[i]) dist[i] = nd;
        }
        __syncthreads();
    }
}

// ---------------------------------------------------------------------------
// Edge features: row (b,q,k) = [ x_i (di) | x_j - x_i (di) | 0 pad ]  -> f16
// ---------------------------------------------------------------------------
__global__ void edge_feat(const float* __restrict__ x,
                          const int* __restrict__ idx,
                          int Nq, int di, int Kp, int rowsTotal,
                          _Float16* __restrict__ Aout)
{
    const size_t gid = (size_t)blockIdx.x * blockDim.x + threadIdx.x;
    const size_t total = (size_t)rowsTotal * Kp;
    if (gid >= total) return;
    const int    c   = (int)(gid % Kp);
    const size_t row = gid / Kp;
    const int    k   = (int)(row % 16);
    const size_t bq  = row / 16;           // b*Nq + q
    float v = 0.f;
    if (c < 2 * di) {
        const float* xi = x + bq * di;
        if (c < di) v = xi[c];
        else {
            const int    nb = idx[bq * 16 + k];
            const size_t bb = bq / Nq;
            v = x[((size_t)bb * Nq + nb) * di + (c - di)] - xi[c - di];
        }
    }
    Aout[gid] = (_Float16)v;
}

// ---------------------------------------------------------------------------
// Batch-norm statistics: one block per channel (population mean / var).
// ---------------------------------------------------------------------------
__global__ void bn_stats(const float* __restrict__ H, int M, int N,
                         float* __restrict__ mv)
{
    const int n = blockIdx.x;
    const int t = threadIdx.x;
    const int T = blockDim.x;
    float s = 0.f, s2 = 0.f;
    for (int m = t; m < M; m += T) {
        const float v = H[(size_t)m * N + n];
        s += v; s2 += v * v;
    }
    __shared__ float sh[256], sh2[256];
    sh[t] = s; sh2[t] = s2;
    __syncthreads();
    for (int st = T >> 1; st > 0; st >>= 1) {
        if (t < st) { sh[t] += sh[t + st]; sh2[t] += sh2[t + st]; }
        __syncthreads();
    }
    if (t == 0) {
        const float mean = sh[0] / (float)M;
        mv[n]     = mean;
        mv[N + n] = fmaxf(sh2[0] / (float)M - mean * mean, 0.f);
    }
}

__global__ void bn_relu_f16(const float* __restrict__ H, const float* __restrict__ mv,
                            const float* __restrict__ g, const float* __restrict__ be,
                            size_t total, int N, _Float16* __restrict__ out)
{
    const size_t gid = (size_t)blockIdx.x * blockDim.x + threadIdx.x;
    if (gid >= total) return;
    const int n = (int)(gid % N);
    const float v = (H[gid] - mv[n]) * rsqrtf(mv[N + n] + 1e-5f) * g[n] + be[n];
    out[gid] = (_Float16)fmaxf(v, 0.f);
}

__global__ void bn_relu_f32(const float* __restrict__ H, const float* __restrict__ mv,
                            const float* __restrict__ g, const float* __restrict__ be,
                            size_t total, int N, float* __restrict__ out)
{
    const size_t gid = (size_t)blockIdx.x * blockDim.x + threadIdx.x;
    if (gid >= total) return;
    const int n = (int)(gid % N);
    const float v = (H[gid] - mv[n]) * rsqrtf(mv[N + n] + 1e-5f) * g[n] + be[n];
    out[gid] = fmaxf(v, 0.f);
}

// bn + relu + max over 16 consecutive rows (EdgeConv neighbor reduction)
__global__ void bn_relu_max16(const float* __restrict__ H, const float* __restrict__ mv,
                              const float* __restrict__ g, const float* __restrict__ be,
                              int RowsOut, int N, float* __restrict__ out)
{
    const size_t gid = (size_t)blockIdx.x * blockDim.x + threadIdx.x;
    if (gid >= (size_t)RowsOut * N) return;
    const int    n = (int)(gid % N);
    const size_t r = gid / N;
    const float mean = mv[n];
    const float inv  = rsqrtf(mv[N + n] + 1e-5f);
    const float gg = g[n], bb = be[n];
    const float* base = H + (r * 16) * (size_t)N + n;
    float best = -3.4e38f;
#pragma unroll
    for (int k = 0; k < 16; ++k) {
        float v = (base[(size_t)k * N] - mean) * inv * gg + bb;
        best = fmaxf(best, fmaxf(v, 0.f));
    }
    out[gid] = best;
}

// ---------------------------------------------------------------------------
// Row gather: dst[b,q,:] = src[b, idx[b,q], :]
// ---------------------------------------------------------------------------
__global__ void gather_rows(const float* __restrict__ src, int Ns, int Cdim,
                            const int* __restrict__ idx, int Nq,
                            float* __restrict__ dst)
{
    const size_t gid = (size_t)blockIdx.x * blockDim.x + threadIdx.x;
    const size_t total = (size_t)B * Nq * Cdim;
    if (gid >= total) return;
    const int    c  = (int)(gid % Cdim);
    const size_t bq = gid / Cdim;
    const int    q  = (int)(bq % Nq);
    const int    b  = (int)(bq / Nq);
    const int    s  = idx[(size_t)b * Nq + q];
    dst[gid] = src[((size_t)b * Ns + s) * Cdim + c];
}

// ---------------------------------------------------------------------------
// KNN-3 inverse-distance interpolation.
// ---------------------------------------------------------------------------
__global__ void knn_interp(const float* __restrict__ xsrc,
                           const int* __restrict__ idx,
                           const float* __restrict__ d2,
                           int Nq, int Ns, int Cdim,
                           float* __restrict__ out)
{
    const size_t gid = (size_t)blockIdx.x * blockDim.x + threadIdx.x;
    const size_t total = (size_t)B * Nq * Cdim;
    if (gid >= total) return;
    const int    c  = (int)(gid % Cdim);
    const size_t bq = gid / Cdim;
    const int    b  = (int)(bq / Nq);
    float ws = 0.f, acc = 0.f;
#pragma unroll
    for (int j = 0; j < 3; ++j) {
        const float w = 1.f / fmaxf(d2[bq * 3 + j], 1e-16f);
        const int   s = idx[bq * 3 + j];
        acc += w * xsrc[((size_t)b * Ns + s) * Cdim + c];
        ws  += w;
    }
    out[gid] = acc / ws;
}

// ---------------------------------------------------------------------------
// Concatenate two f32 feature maps into a zero-padded f16 GEMM A buffer.
// ---------------------------------------------------------------------------
__global__ void concat2_f16(const float* __restrict__ X0, int C0,
                            const float* __restrict__ X1, int C1,
                            size_t Rows, int Kp, _Float16* __restrict__ Aout)
{
    const size_t gid = (size_t)blockIdx.x * blockDim.x + threadIdx.x;
    if (gid >= Rows * (size_t)Kp) return;
    const int    c = (int)(gid % Kp);
    const size_t r = gid / Kp;
    float v = 0.f;
    if (c < C0)            v = X0[r * C0 + c];
    else if (c < C0 + C1)  v = X1[r * C1 + (c - C0)];
    Aout[gid] = (_Float16)v;
}

// Convert f32 weight (K x N) to zero-padded TRANSPOSED f16 (Np x Kp, n-major)
__global__ void conv_w_f16t(const float* __restrict__ Win, int K, int N, int Kp, int Np,
                            _Float16* __restrict__ out)
{
    const size_t gid = (size_t)blockIdx.x * blockDim.x + threadIdx.x;
    if (gid >= (size_t)Np * Kp) return;
    const int k = (int)(gid % Kp);
    const int n = (int)(gid / Kp);
    out[gid] = (_Float16)((k < K && n < N) ? Win[(size_t)k * N + n] : 0.f);
}

__global__ void pad_bias(const float* __restrict__ bsrc, int N, int Np,
                         float* __restrict__ out)
{
    const int i = blockIdx.x * blockDim.x + threadIdx.x;
    if (i < Np) out[i] = (i < N) ? bsrc[i] : 0.f;
}

// log_softmax over 13 classes (stored with stride 16) -> contiguous stride 13
__global__ void log_softmax13(const float* __restrict__ H, int Rows,
                              float* __restrict__ out)
{
    const int r = blockIdx.x * blockDim.x + threadIdx.x;
    if (r >= Rows) return;
    const float* h = H + (size_t)r * 16;
    float mx = -3.4e38f;
#pragma unroll
    for (int c = 0; c < 13; ++c) mx = fmaxf(mx, h[c]);
    float s = 0.f;
#pragma unroll
    for (int c = 0; c < 13; ++c) s += expf(h[c] - mx);
    const float ls = logf(s);
#pragma unroll
    for (int c = 0; c < 13; ++c) out[(size_t)r * 13 + c] = h[c] - mx - ls;
}

// ---------------------------------------------------------------------------
// Host driver
// ---------------------------------------------------------------------------
struct ConvRef { const float *w1, *b1, *g1, *be1, *w2, *b2, *g2, *be2; };
struct DecRef  { const float *w, *b, *g, *be; };
struct HeadRef { const float *w1, *b1, *g, *be, *w2, *b2; };
struct ConvW   { const _Float16 *w1, *w2; const float *b1, *b2, *g1, *g2, *be1, *be2; };

extern "C" void kernel_launch(void* const* d_in, const int* in_sizes, int n_in,
                              void* d_out, int out_size, void* d_ws, size_t ws_size,
                              hipStream_t stream)
{
    (void)out_size; (void)ws_size;
    // ---- Input layout detection (insertion order vs jax sorted pytree) ----
    const bool insertion = (in_sizes[0] == B * 4096 * 6);
    int ix, ipos, pbase;
    if (insertion) { ix = 0; ipos = 1; pbase = 2; }
    else           { ix = n_in - 1; ipos = n_in - 2; pbase = 0; }
    const float* x   = (const float*)d_in[ix];
    const float* pos = (const float*)d_in[ipos];
    auto P = [&](int i) { return (const float*)d_in[pbase + i]; };

    ConvRef conv[4]; DecRef dec[3]; HeadRef head;
    for (int c = 0; c < 4; ++c) {
        const int b0 = c * 8;
        if (insertion) // w1,b1,g1,be1,w2,b2,g2,be2
            conv[c] = { P(b0+0), P(b0+1), P(b0+2), P(b0+3), P(b0+4), P(b0+5), P(b0+6), P(b0+7) };
        else           // sorted: b1,b2,be1,be2,g1,g2,w1,w2
            conv[c] = { P(b0+6), P(b0+0), P(b0+4), P(b0+2), P(b0+7), P(b0+1), P(b0+5), P(b0+3) };
    }
    for (int d = 0; d < 3; ++d) {
        const int b0 = 32 + d * 4;
        if (insertion) dec[d] = { P(b0+0), P(b0+1), P(b0+2), P(b0+3) };      // w,b,g,be
        else           dec[d] = { P(b0+3), P(b0+0), P(b0+2), P(b0+1) };      // b,be,g,w
    }
    {
        const int b0 = 44;
        if (insertion) head = { P(b0+0), P(b0+1), P(b0+2), P(b0+3), P(b0+4), P(b0+5) };
        else           head = { P(b0+4), P(b0+0), P(b0+3), P(b0+2), P(b0+5), P(b0+1) };
    }

    // ---- Workspace bump allocator ----
    char*  wsb = (char*)d_ws;
    size_t off = 0;
    auto alloc = [&](size_t bytes) -> void* {
        void* p = wsb + off;
        off = (off + bytes + 255) & ~(size_t)255;
        return p;
    };
    // Big ping-pong scratch
    _Float16* SA = (_Float16*)alloc((size_t)16777216 * 2);   // f16 A operands
    float*    SH = (float*)   alloc((size_t)16777216 * 4);   // f32 GEMM outputs
    float*    mv = (float*)   alloc(1024 * 4);
    int*   idx16 = (int*)     alloc((size_t)B * 4096 * 16 * 4);
    int*   idx3  = (int*)     alloc((size_t)B * 4096 * 3 * 4);
    float* d23   = (float*)   alloc((size_t)B * 4096 * 3 * 4);
    int*   ifps  = (int*)     alloc(4096 * 4);
    // Persistent activations
    float* x1  = (float*)alloc((size_t)B * 4096 * 64 * 4);
    float* x1s = (float*)alloc((size_t)B * 1024 * 64 * 4);
    float* x2  = (float*)alloc((size_t)B * 1024 * 128 * 4);
    float* x2s = (float*)alloc((size_t)B * 256 * 128 * 4);
    float* x3  = (float*)alloc((size_t)B * 256 * 256 * 4);
    float* x3s = (float*)alloc((size_t)B * 64 * 256 * 4);
    float* x4  = (float*)alloc((size_t)B * 64 * 512 * 4);
    float* pos1 = (float*)alloc((size_t)B * 1024 * 3 * 4);
    float* pos2 = (float*)alloc((size_t)B * 256 * 3 * 4);
    float* pos3 = (float*)alloc((size_t)B * 64 * 3 * 4);
    float* up2 = (float*)alloc((size_t)B * 256 * 512 * 4);
    float* up1 = (float*)alloc((size_t)B * 1024 * 256 * 4);
    float* up0 = (float*)alloc((size_t)B * 4096 * 128 * 4);
    float* d2b = (float*)alloc((size_t)B * 256 * 256 * 4);
    float* d1b = (float*)alloc((size_t)B * 1024 * 128 * 4);
    float* d0b = (float*)alloc((size_t)B * 4096 * 64 * 4);
    // f16 transposed weights (zero-padded, N-major)
    _Float16* c1w1h = (_Float16*)alloc((size_t)64 * 32 * 2);
    _Float16* c1w2h = (_Float16*)alloc((size_t)64 * 64 * 2);
    _Float16* c2w1h = (_Float16*)alloc((size_t)128 * 128 * 2);
    _Float16* c2w2h = (_Float16*)alloc((size_t)128 * 128 * 2);
    _Float16* c3w1h = (_Float16*)alloc((size_t)256 * 256 * 2);
    _Float16* c3w2h = (_Float16*)alloc((size_t)256 * 256 * 2);
    _Float16* c4w1h = (_Float16*)alloc((size_t)512 * 512 * 2);
    _Float16* c4w2h = (_Float16*)alloc((size_t)512 * 512 * 2);
    _Float16* d1wh  = (_Float16*)alloc((size_t)256 * 768 * 2);
    _Float16* d2wh  = (_Float16*)alloc((size_t)128 * 384 * 2);
    _Float16* d3wh  = (_Float16*)alloc((size_t)64 * 192 * 2);
    _Float16* hw1h  = (_Float16*)alloc((size_t)64 * 96 * 2);
    _Float16* hw2h  = (_Float16*)alloc((size_t)16 * 64 * 2);
    float*    hb2p  = (float*)alloc(16 * 4);

    // ---- Weight conversion (transpose to N-major, zero pad) ----
    auto cw = [&](const float* Wf, int K, int N, int Kp, int Np, _Float16* outp) {
        const size_t tot = (size_t)Np * Kp;
        conv_w_f16t<<<DIVUP(tot, 256), 256, 0, stream>>>(Wf, K, N, Kp, Np, outp);
    };
    cw(conv[0].w1, 12, 64, 32, 64, c1w1h);     cw(conv[0].w2, 64, 64, 64, 64, c1w2h);
    cw(conv[1].w1, 128, 128, 128, 128, c2w1h); cw(conv[1].w2, 128, 128, 128, 128, c2w2h);
    cw(conv[2].w1, 256, 256, 256, 256, c3w1h); cw(conv[2].w2, 256, 256, 256, 256, c3w2h);
    cw(conv[3].w1, 512, 512, 512, 512, c4w1h); cw(conv[3].w2, 512, 512, 512, 512, c4w2h);
    cw(dec[0].w, 768, 256, 768, 256, d1wh);
    cw(dec[1].w, 384, 128, 384, 128, d2wh);
    cw(dec[2].w, 192, 64, 192, 64, d3wh);
    cw(head.w1, 70, 64, 96, 64, hw1h);
    cw(head.w2, 64, 13, 64, 16, hw2h);
    pad_bias<<<1, 32, 0, stream>>>(head.b2, 13, 16, hb2p);

    auto launch_gemm = [&](const _Float16* Am, const _Float16* Wm, const float* bm,
                           float* Cm, int M, int Kp, int N) {
        if (N % 64 == 0) {
            dim3 g(DIVUP(M / 16, 4), N / 64);
            gemm_f16_wmma<4><<<g, 128, 0, stream>>>(Am, Wm, bm, Cm, M, Kp, N);
        } else {
            dim3 g(DIVUP(M / 16, 4), N / 16);
            gemm_f16_wmma<1><<<g, 128, 0, stream>>>(Am, Wm, bm, Cm, M, Kp, N);
        }
    };

    // ---- EdgeConv block ----
    auto run_conv = [&](const float* xin, const float* posin, int Nq, int di, int dout,
                        const ConvW& W, float* xout) {
        dim3 kg(DIVUP(Nq, 128), B);
        knn_kernel<16, true><<<kg, 128, 0, stream>>>(posin, posin, Nq, Nq, idx16,
                                                     (float*)nullptr);
        const int rows = B * Nq * 16;
        const int Kp = ((2 * di + 31) / 32) * 32;
        const size_t totA = (size_t)rows * Kp;
        edge_feat<<<DIVUP(totA, 256), 256, 0, stream>>>(xin, idx16, Nq, di, Kp, rows, SA);
        launch_gemm(SA, W.w1, W.b1, SH, rows, Kp, dout);
        bn_stats<<<dout, 256, 0, stream>>>(SH, rows, dout, mv);
        const size_t tot1 = (size_t)rows * dout;
        bn_relu_f16<<<DIVUP(tot1, 256), 256, 0, stream>>>(SH, mv, W.g1, W.be1, tot1, dout, SA);
        launch_gemm(SA, W.w2, W.b2, SH, rows, dout, dout);
        bn_stats<<<dout, 256, 0, stream>>>(SH, rows, dout, mv);
        const size_t tot2 = (size_t)B * Nq * dout;
        bn_relu_max16<<<DIVUP(tot2, 256), 256, 0, stream>>>(SH, mv, W.g2, W.be2,
                                                            B * Nq, dout, xout);
    };

    auto run_interp = [&](const float* posq, int Nq, const float* poss, int Ns,
                          const float* xsrc, int Cdim, float* outb) {
        dim3 kg(DIVUP(Nq, 128), B);
        knn_kernel<3, false><<<kg, 128, 0, stream>>>(posq, poss, Nq, Ns, idx3, d23);
        const size_t tot = (size_t)B * Nq * Cdim;
        knn_interp<<<DIVUP(tot, 256), 256, 0, stream>>>(xsrc, idx3, d23, Nq, Ns, Cdim, outb);
    };

    auto run_dec = [&](const float* upb, int Cup, const float* skip, int Cskip, int Rows,
                       const _Float16* Wm, const DecRef& dr, int dout, float* outb) {
        const int Kp = Cup + Cskip;  // multiples of 32 by construction
        const size_t tot = (size_t)Rows * Kp;
        concat2_f16<<<DIVUP(tot, 256), 256, 0, stream>>>(upb, Cup, skip, Cskip,
                                                         (size_t)Rows, Kp, SA);
        launch_gemm(SA, Wm, dr.b, SH, Rows, Kp, dout);
        bn_stats<<<dout, 256, 0, stream>>>(SH, Rows, dout, mv);
        const size_t t2 = (size_t)Rows * dout;
        bn_relu_f32<<<DIVUP(t2, 256), 256, 0, stream>>>(SH, mv, dr.g, dr.be, t2, dout, outb);
    };

    // ---- Encoder ----
    const ConvW cw1 = { c1w1h, c1w2h, conv[0].b1, conv[0].b2, conv[0].g1, conv[0].g2, conv[0].be1, conv[0].be2 };
    const ConvW cw2 = { c2w1h, c2w2h, conv[1].b1, conv[1].b2, conv[1].g1, conv[1].g2, conv[1].be1, conv[1].be2 };
    const ConvW cw3 = { c3w1h, c3w2h, conv[2].b1, conv[2].b2, conv[2].g1, conv[2].g2, conv[2].be1, conv[2].be2 };
    const ConvW cw4 = { c4w1h, c4w2h, conv[3].b1, conv[3].b2, conv[3].g1, conv[3].g2, conv[3].be1, conv[3].be2 };

    run_conv(x, pos, 4096, 6, 64, cw1, x1);

    fps_kernel<<<B, 256, 0, stream>>>(pos, 4096, 1024, ifps);
    gather_rows<<<DIVUP((size_t)B * 1024 * 3, 256), 256, 0, stream>>>(pos, 4096, 3, ifps, 1024, pos1);
    gather_rows<<<DIVUP((size_t)B * 1024 * 64, 256), 256, 0, stream>>>(x1, 4096, 64, ifps, 1024, x1s);
    run_conv(x1s, pos1, 1024, 64, 128, cw2, x2);

    fps_kernel<<<B, 256, 0, stream>>>(pos1, 1024, 256, ifps);
    gather_rows<<<DIVUP((size_t)B * 256 * 3, 256), 256, 0, stream>>>(pos1, 1024, 3, ifps, 256, pos2);
    gather_rows<<<DIVUP((size_t)B * 256 * 128, 256), 256, 0, stream>>>(x2, 1024, 128, ifps, 256, x2s);
    run_conv(x2s, pos2, 256, 128, 256, cw3, x3);

    fps_kernel<<<B, 256, 0, stream>>>(pos2, 256, 64, ifps);
    gather_rows<<<DIVUP((size_t)B * 64 * 3, 256), 256, 0, stream>>>(pos2, 256, 3, ifps, 64, pos3);
    gather_rows<<<DIVUP((size_t)B * 64 * 256, 256), 256, 0, stream>>>(x3, 256, 256, ifps, 64, x3s);
    run_conv(x3s, pos3, 64, 256, 512, cw4, x4);

    // ---- Decoder ----
    run_interp(pos2, 256, pos3, 64, x4, 512, up2);
    run_dec(up2, 512, x3, 256, B * 256, d1wh, dec[0], 256, d2b);

    run_interp(pos1, 1024, pos2, 256, d2b, 256, up1);
    run_dec(up1, 256, x2, 128, B * 1024, d2wh, dec[1], 128, d1b);

    run_interp(pos, 4096, pos1, 1024, d1b, 128, up0);
    run_dec(up0, 128, x1, 64, B * 4096, d3wh, dec[2], 64, d0b);

    // ---- Head ----
    {
        const int Rows = B * 4096;
        const size_t tot = (size_t)Rows * 96;
        concat2_f16<<<DIVUP(tot, 256), 256, 0, stream>>>(d0b, 64, x, 6, (size_t)Rows, 96, SA);
        launch_gemm(SA, hw1h, head.b1, SH, Rows, 96, 64);
        bn_stats<<<64, 256, 0, stream>>>(SH, Rows, 64, mv);
        const size_t t2 = (size_t)Rows * 64;
        bn_relu_f16<<<DIVUP(t2, 256), 256, 0, stream>>>(SH, mv, head.g, head.be, t2, 64, SA);
        launch_gemm(SA, hw2h, hb2p, SH, Rows, 64, 16);
        log_softmax13<<<DIVUP(Rows, 256), 256, 0, stream>>>(SH, Rows, (float*)d_out);
    }
}
